// AttentionHead_37469294691171
// MI455X (gfx1250) — matrix operand order repaired and checked by reading, not compiled, not run on hardware
//
#include <hip/hip_runtime.h>
#include <hip/hip_bf16.h>
#include <math.h>

// Problem constants (match reference)
#define BB 8
#define TT 4096
#define EE 2048
#define DD 128
#define NROW (BB * TT)       // 32768 global rows
#define NT (TT / 16)         // 256 16-row tiles per sequence
#define SCALE 0.08838834764831843f  // 1/sqrt(128)

typedef __attribute__((ext_vector_type(16))) _Float16 v16h;
typedef __attribute__((ext_vector_type(8)))  _Float16 v8h;
typedef __attribute__((ext_vector_type(4)))  _Float16 v4h;
typedef __attribute__((ext_vector_type(8)))  float    v8f;
typedef __attribute__((ext_vector_type(4)))  float    v4f;

// ---------------------------------------------------------------------------
// WMMA fragment loaders (wave32, 16x16x32 f16)
//
// A-matrix 16x32 (MxK), ISA 7.12.2: lanes 0-15 -> M=lane, halves 0-7 = K0..7,
// halves 8-15 = K16..23; lanes 16-31 -> M=lane-16, halves 0-7 = K8..15,
// halves 8-15 = K24..31.  src is row-major [M][K] with leading dim ld.
__device__ __forceinline__ v16h load_a_frag(const _Float16* src, int ld, int lane) {
    int row = lane & 15;
    int k0  = (lane & 16) ? 8 : 0;
    const _Float16* p = src + (long)row * ld + k0;
    v8h lo = *reinterpret_cast<const v8h*>(p);
    v8h hi = *reinterpret_cast<const v8h*>(p + 16);
    v16h r;
#pragma unroll
    for (int i = 0; i < 8; ++i) { r[i] = lo[i]; r[i + 8] = hi[i]; }
    return r;
}

// B-matrix 32x16 (KxN), matching the A layout: lanes 0-15 -> N=lane, halves 0-7 = K0..7,
// halves 8-15 = K16..23; lanes 16-31 -> N=lane-16, halves 0-7 = K8..15, halves 8-15 = K24..31.
// src is N-major [N][K] (i.e. B transposed, row-major) with leading dim ld.
__device__ __forceinline__ v16h load_b_frag(const _Float16* src, int ld, int lane) {
    int col = lane & 15;
    int k0  = (lane & 16) ? 8 : 0;
    const _Float16* p = src + (long)col * ld + k0;
    v8h lo = *reinterpret_cast<const v8h*>(p);
    v8h hi = *reinterpret_cast<const v8h*>(p + 16);
    v16h r;
#pragma unroll
    for (int i = 0; i < 8; ++i) { r[i] = lo[i]; r[i + 8] = hi[i]; }
    return r;
}

__device__ __forceinline__ v8f wmma_f16(v16h a, v16h b, v8f c) {
    return __builtin_amdgcn_wmma_f32_16x16x32_f16(false, a, false, b,
                                                  (short)0, c, false, false);
}

__device__ __forceinline__ void merge_stats(float& m, float& z, float m2, float z2) {
    float mn = fmaxf(m, m2);
    z = z * __expf(m - mn) + z2 * __expf(m2 - mn);
    m = mn;
}

// ---------------------------------------------------------------------------
// Kernel 0: W [E][128] f32 -> Wt [128][E] f16 (transpose + convert)
__global__ __launch_bounds__(256) void wcvt_kernel(const float* __restrict__ W,
                                                   _Float16* __restrict__ Wt) {
    int idx = blockIdx.x * 256 + threadIdx.x;     // E*D elements
    int e = idx >> 7;
    int d = idx & 127;
    Wt[(long)d * EE + e] = (_Float16)W[(long)e * DD + d];
}

// ---------------------------------------------------------------------------
// Kernel 1: fused QKV projection.  Each block: 32 rows of x, 8 waves.
// Wave w owns output columns [16w,16w+16) of q,k,v and both 16-row M-tiles,
// so its 3 B-fragments are reused by 2 A-fragments (6 WMMAs per K-step).
__global__ __launch_bounds__(256) void qkv_proj_kernel(
    const float* __restrict__ x,
    const _Float16* __restrict__ Wtq, const _Float16* __restrict__ Wtk,
    const _Float16* __restrict__ Wtv,
    _Float16* __restrict__ qf, _Float16* __restrict__ kf,
    _Float16* __restrict__ vT) {
    __shared__ _Float16 xs[32 * 40];   // 32x32 f16 tile, stride 40 (bank-clean)
    const int tid  = threadIdx.x;
    const int wave = tid >> 5;
    const int lane = tid & 31;
    const long row0 = (long)blockIdx.x * 32;

    v8f aq0 = {}, aq1 = {}, ak0 = {}, ak1 = {}, av0 = {}, av1 = {};
    const long wcol = (long)wave * 16;

    // staging coordinates: 32 rows x 32 cols, 256 threads x float4
    const int srow = tid >> 3;          // 0..31
    const int scol = (tid & 7) * 4;     // 0,4,...,28
    const float* xrow = x + (row0 + srow) * (long)EE + scol;

    for (int e0 = 0; e0 < EE; e0 += 32) {
        // vectorized convert-load: global_load_b128 -> packed cvt -> ds_store_b64
        v4f xv = *reinterpret_cast<const v4f*>(xrow + e0);
        v4h hv;
#pragma unroll
        for (int i = 0; i < 4; ++i) hv[i] = (_Float16)xv[i];
        *reinterpret_cast<v4h*>(&xs[srow * 40 + scol]) = hv;
        if (e0 + 32 < EE)
            __builtin_prefetch(xrow + e0 + 32, 0, 1);   // global_prefetch_b8
        __syncthreads();

        v16h a0 = load_a_frag(xs, 40, lane);            // rows 0..15
        v16h a1 = load_a_frag(xs + 16 * 40, 40, lane);  // rows 16..31
        v16h bq = load_b_frag(Wtq + wcol * EE + e0, EE, lane);
        v16h bk = load_b_frag(Wtk + wcol * EE + e0, EE, lane);
        v16h bv = load_b_frag(Wtv + wcol * EE + e0, EE, lane);
        aq0 = wmma_f16(a0, bq, aq0);
        aq1 = wmma_f16(a1, bq, aq1);
        ak0 = wmma_f16(a0, bk, ak0);
        ak1 = wmma_f16(a1, bk, ak1);
        av0 = wmma_f16(a0, bv, av0);
        av1 = wmma_f16(a1, bv, av1);
        __syncthreads();
    }

    // C layout: VGPR r -> (M = r + 8*(lane>=16), N = lane&15)
    const int col   = wave * 16 + (lane & 15);
    const int rbase = (lane & 16) ? 8 : 0;
#pragma unroll
    for (int r = 0; r < 8; ++r) {
        long row = row0 + rbase + r;
        qf[row * DD + col] = (_Float16)aq0[r];
        kf[row * DD + col] = (_Float16)ak0[r];
        vT[(long)col * NROW + row] = (_Float16)av0[r];   // V stored transposed
        long row2 = row + 16;
        qf[row2 * DD + col] = (_Float16)aq1[r];
        kf[row2 * DD + col] = (_Float16)ak1[r];
        vT[(long)col * NROW + row2] = (_Float16)av1[r];
    }
}

// ---------------------------------------------------------------------------
// Kernel 2: per-key-column softmax stats (softmax over the QUERY axis).
// Block (jt, b): owns 16 key columns j0..j0+15.  Waves stride over query
// tiles i >= j keeping online (max, sumexp) per column; combine at the end.
__global__ __launch_bounds__(256) void col_stats_kernel(
    const _Float16* __restrict__ qf, const _Float16* __restrict__ kf,
    float* __restrict__ mOut, float* __restrict__ zOut) {
    const int jt = blockIdx.x;
    const int b  = blockIdx.y;
    const int j0 = jt * 16;
    const int tid  = threadIdx.x;
    const int wave = tid >> 5;
    const int lane = tid & 31;
    const _Float16* qB = qf + (long)b * TT * DD;
    const _Float16* kB = kf + (long)b * TT * DD;

    // K columns fixed for the block: preload B fragments (K-dim = D = 4x32)
    v16h bk[4];
#pragma unroll
    for (int kk = 0; kk < 4; ++kk)
        bk[kk] = load_b_frag(kB + (long)j0 * DD + kk * 32, DD, lane);

    const float NEG = -__builtin_inff();
    float m_run = NEG, z_run = 0.f;
    const int jcol  = j0 + (lane & 15);
    const int rbase = (lane & 16) ? 8 : 0;

    for (int it = jt + wave; it < NT; it += 8) {
        const int i0 = it * 16;
        v8f s = {};
#pragma unroll
        for (int kk = 0; kk < 4; ++kk) {
            v16h a = load_a_frag(qB + (long)i0 * DD + kk * 32, DD, lane);
            s = wmma_f16(a, bk[kk], s);
        }
        // mask + tile max (this lane's 8 rows)
        float sv[8];
        float mloc = NEG;
#pragma unroll
        for (int r = 0; r < 8; ++r) {
            int i = i0 + rbase + r;
            float val = (i >= jcol) ? s[r] * SCALE : NEG;
            sv[r] = val;
            mloc = fmaxf(mloc, val);
        }
        float mt = fmaxf(mloc, __shfl_xor(mloc, 16, 32));  // lane pair = same column
        float se = 0.f;
#pragma unroll
        for (int r = 0; r < 8; ++r) se += __expf(sv[r] - mt);
        se += __shfl_xor(se, 16, 32);
        merge_stats(m_run, z_run, mt, se);
    }

    __shared__ float smem_m[8][16];
    __shared__ float smem_z[8][16];
    if (lane < 16) { smem_m[wave][lane] = m_run; smem_z[wave][lane] = z_run; }
    __syncthreads();
    if (tid < 16) {
        float M = NEG, Z = 0.f;
#pragma unroll
        for (int w = 0; w < 8; ++w) merge_stats(M, Z, smem_m[w][tid], smem_z[w][tid]);
        long j = (long)b * TT + j0 + tid;
        mOut[j] = M;
        zOut[j] = (Z > 0.f) ? 1.0f / Z : 0.f;   // store 1/Z
    }
}

// ---------------------------------------------------------------------------
// Kernel 3: output pass.  Block (it, b): 16 query rows.  Per chunk of 128
// keys, wave w computes score subtile jt=c0+w, builds p=exp(s-m[j])/Z[j],
// stages p (f16) in LDS, then every wave does p @ V for its 16 output dims.
__global__ __launch_bounds__(256) void attn_out_kernel(
    const _Float16* __restrict__ qf, const _Float16* __restrict__ kf,
    const _Float16* __restrict__ vT,
    const float* __restrict__ mC, const float* __restrict__ ziC,
    float* __restrict__ out) {
    __shared__ _Float16 ps[16 * 136];  // p tile 16x128 f16, stride 136 (bank-clean)
    const int it = blockIdx.x;
    const int b  = blockIdx.y;
    const int i0 = it * 16;
    const int tid  = threadIdx.x;
    const int wave = tid >> 5;
    const int lane = tid & 31;
    const _Float16* qB = qf + (long)b * TT * DD;
    const _Float16* kB = kf + (long)b * TT * DD;
    const int rbase = (lane & 16) ? 8 : 0;

    // query A-fragments reused for every key tile
    v16h aq[4];
#pragma unroll
    for (int kk = 0; kk < 4; ++kk)
        aq[kk] = load_a_frag(qB + (long)i0 * DD + kk * 32, DD, lane);

    v8f acc = {};
    for (int c0 = 0; c0 <= it; c0 += 8) {
        const int jt = c0 + wave;
        const int lcol = wave * 16 + (lane & 15);
        if (jt <= it) {
            const int j0 = jt * 16;
            v8f s = {};
#pragma unroll
            for (int kk = 0; kk < 4; ++kk) {
                v16h bk = load_b_frag(kB + (long)j0 * DD + kk * 32, DD, lane);
                s = wmma_f16(aq[kk], bk, s);
            }
            const int j = j0 + (lane & 15);
            const float mj = mC[(long)b * TT + j];
            const float zi = ziC[(long)b * TT + j];
#pragma unroll
            for (int r = 0; r < 8; ++r) {
                int i = i0 + rbase + r;
                float p = (i >= j) ? __expf(s[r] * SCALE - mj) * zi : 0.f;
                ps[(rbase + r) * 136 + lcol] = (_Float16)p;
            }
        } else {
#pragma unroll
            for (int r = 0; r < 8; ++r)
                ps[(rbase + r) * 136 + lcol] = (_Float16)0.f;
        }
        __syncthreads();

        // acc += p[16x128] @ V[128x16] for this wave's 16 output dims
        const long jbase = (long)b * TT + (long)c0 * 16;
#pragma unroll
        for (int kk = 0; kk < 4; ++kk) {
            v16h ap = load_a_frag(ps + kk * 32, 136, lane);
            v16h bv = load_b_frag(vT + (long)wave * 16 * NROW + jbase + kk * 32,
                                  NROW, lane);
            acc = wmma_f16(ap, bv, acc);
        }
        __syncthreads();
    }

    const int dcol = wave * 16 + (lane & 15);
#pragma unroll
    for (int r = 0; r < 8; ++r)
        out[((long)b * TT + i0 + rbase + r) * DD + dcol] = acc[r];
}

// ---------------------------------------------------------------------------
extern "C" void kernel_launch(void* const* d_in, const int* in_sizes, int n_in,
                              void* d_out, int out_size, void* d_ws, size_t ws_size,
                              hipStream_t stream) {
    (void)in_sizes; (void)n_in; (void)out_size; (void)ws_size;
    const float* x  = (const float*)d_in[0];
    const float* Wq = (const float*)d_in[1];
    const float* Wk = (const float*)d_in[2];
    const float* Wv = (const float*)d_in[3];
    float* out = (float*)d_out;

    // workspace layout (all pointers 16B aligned)
    _Float16* Wtq = (_Float16*)d_ws;                    // [128][E]
    _Float16* Wtk = Wtq + (size_t)DD * EE;
    _Float16* Wtv = Wtk + (size_t)DD * EE;
    _Float16* qf  = Wtv + (size_t)DD * EE;              // [B*T][128]
    _Float16* kf  = qf  + (size_t)NROW * DD;
    _Float16* vT  = kf  + (size_t)NROW * DD;            // [128][B*T]
    float*    mC  = (float*)(vT + (size_t)DD * NROW);   // [B*T]
    float*    ziC = mC + NROW;

    // 0) transpose+convert weights to f16 (L2-resident, reused by all blocks)
    const int wblk = (EE * DD) / 256;
    wcvt_kernel<<<wblk, 256, 0, stream>>>(Wq, Wtq);
    wcvt_kernel<<<wblk, 256, 0, stream>>>(Wk, Wtk);
    wcvt_kernel<<<wblk, 256, 0, stream>>>(Wv, Wtv);

    // 1) fused QKV projection (32-row M-tiles, B-fragment reuse)
    qkv_proj_kernel<<<NROW / 32, 256, 0, stream>>>(x, Wtq, Wtk, Wtv, qf, kf, vT);

    // 2) per-key-column softmax stats (softmax over query axis)
    col_stats_kernel<<<dim3(NT, BB), 256, 0, stream>>>(qf, kf, mC, ziC);

    // 3) recompute scores, normalize per column, P @ V
    attn_out_kernel<<<dim3(NT, BB), 256, 0, stream>>>(qf, kf, vT, mC, ziC, out);
}
